// InhibitorAttention_55817394979296
// MI455X (gfx1250) — matrix operand order repaired and checked by reading, not compiled, not run on hardware
//
#include <hip/hip_runtime.h>

#define S  512
#define DD 512
#define NH 8
#define DH 64

typedef __attribute__((ext_vector_type(16))) _Float16 v16h;
typedef __attribute__((ext_vector_type(8)))  _Float16 v8h;
typedef __attribute__((ext_vector_type(8)))  float    v8f;

// ---------------------------------------------------------------------------
// Conversion kernels
// ---------------------------------------------------------------------------
__global__ __launch_bounds__(256) void cvt_f32_to_f16_kernel(
    const float* __restrict__ x, _Float16* __restrict__ y, int n) {
  int i = blockIdx.x * 256 + threadIdx.x;
  if (i < n) y[i] = (_Float16)x[i];
}

// Wt[n*512 + k] = (f16) W[k*512 + n]   (W stored [in,out] -> Wt stored [out,in])
__global__ __launch_bounds__(256) void cvt_transpose_w_kernel(
    const float* __restrict__ W, _Float16* __restrict__ Wt) {
  int idx = blockIdx.x * 256 + threadIdx.x;  // idx = n*512 + k
  int n = idx >> 9;
  int k = idx & 511;
  Wt[idx] = (_Float16)W[k * DD + n];
}

// ---------------------------------------------------------------------------
// WMMA 16x16 f16 GEMM tile:  C[16x16] = A[16xK] * Bt[16(N) x K]^T  (f32 acc)
// A fragment per lane (16-bit A 16x32 ISA layout): M = lane&15,
//   K runs: [hh*8 .. hh*8+7] and [16+hh*8 .. 16+hh*8+7], hh = lane>>4.
// B fragment per lane (16-bit B 32x16 ISA layout): N = lane&15,
//   K runs: [hh*8 .. hh*8+7] and [16+hh*8 .. 16+hh*8+7] in Bt row (Bt is [N,K]).
// ---------------------------------------------------------------------------
__device__ __forceinline__ v8f wmma_tile_f16(const _Float16* __restrict__ A,
                                             const _Float16* __restrict__ Bt,
                                             int tile_m, int tile_n, int lane) {
  const int hh   = lane >> 4;    // 0 or 1
  const int mrow = lane & 15;
  const _Float16* arow = A  + (size_t)(tile_m * 16 + mrow) * DD + hh * 8;
  const _Float16* brow = Bt + (size_t)(tile_n * 16 + mrow) * DD + hh * 8;
  v8f c = {};
  for (int kb = 0; kb < DD; kb += 32) {
    __builtin_prefetch(arow + kb + 64, 0, 1);  // global_prefetch_b8
    __builtin_prefetch(brow + kb + 64, 0, 1);
    v8h a0 = *(const v8h*)(arow + kb);
    v8h a1 = *(const v8h*)(arow + kb + 16);
    v8h b0 = *(const v8h*)(brow + kb);
    v8h b1 = *(const v8h*)(brow + kb + 16);
    v16h a, b;
#pragma unroll
    for (int e = 0; e < 8; ++e) {
      a[e] = a0[e]; a[e + 8] = a1[e];
      b[e] = b0[e]; b[e + 8] = b1[e];
    }
    // 8 args: (neg_a, A, neg_b, B, c_mod, C, reuse_a, reuse_b)
    c = __builtin_amdgcn_wmma_f32_16x16x32_f16(false, a, false, b,
                                               (short)0, c, false, false);
  }
  return c;
}

// C/D layout: VGPR r -> M = hh*8 + r, N = lane&15
__device__ __forceinline__ void store_tile(float* __restrict__ C,
                                           const float* __restrict__ bias,
                                           v8f c, int tile_m, int tile_n,
                                           int lane) {
  const int hh  = lane >> 4;
  const int col = tile_n * 16 + (lane & 15);
  const float bb = bias[col];
#pragma unroll
  for (int r = 0; r < 8; ++r) {
    int row = tile_m * 16 + hh * 8 + r;
    C[(size_t)row * DD + col] = c[r] + bb;
  }
}

// Fused Q/K/V projection: blockIdx.y selects {Wq,bq,Q}/{Wk,bk,K}/{Wv,bv,V}.
// 256 threads = 8 waves; each wave owns one 16x16 output tile.
__global__ __launch_bounds__(256) void gemm_qkv_kernel(
    const _Float16* __restrict__ Xh,
    const _Float16* __restrict__ Wqt, const _Float16* __restrict__ Wkt,
    const _Float16* __restrict__ Wvt,
    const float* __restrict__ bq, const float* __restrict__ bk,
    const float* __restrict__ bv_,
    float* __restrict__ Q, float* __restrict__ K, float* __restrict__ V) {
  const int lane = threadIdx.x & 31;
  const int wave = threadIdx.x >> 5;
  const int tile = blockIdx.x * 8 + wave;   // 0..1023
  const int tile_m = tile >> 5;
  const int tile_n = tile & 31;

  const _Float16* Bt;
  const float* bias;
  float* C;
  if (blockIdx.y == 0)      { Bt = Wqt; bias = bq;  C = Q; }
  else if (blockIdx.y == 1) { Bt = Wkt; bias = bk;  C = K; }
  else                      { Bt = Wvt; bias = bv_; C = V; }

  v8f c = wmma_tile_f16(Xh, Bt, tile_m, tile_n, lane);
  store_tile(C, bias, c, tile_m, tile_n, lane);
}

// Output projection: out = ctx_h @ Wo + bo
__global__ __launch_bounds__(256) void gemm_out_kernel(
    const _Float16* __restrict__ Ch, const _Float16* __restrict__ Wot,
    const float* __restrict__ bo, float* __restrict__ out) {
  const int lane = threadIdx.x & 31;
  const int wave = threadIdx.x >> 5;
  const int tile = blockIdx.x * 8 + wave;
  const int tile_m = tile >> 5;
  const int tile_n = tile & 31;
  v8f c = wmma_tile_f16(Ch, Wot, tile_m, tile_n, lane);
  store_tile(out, bo, c, tile_m, tile_n, lane);
}

// ---------------------------------------------------------------------------
// Manhattan-distance inhibition core (VALU; not expressible as WMMA).
// Grid: (4 i-blocks of 128 rows, 8 heads). 128 threads, thread t owns row
// i = blk*128+t of its head. q rows staged in LDS with +1 pad (stride 65
// floats -> bank=(t+d)%64, conflict-free); k/v chunks are moved global->LDS
// with the CDNA5 async-to-LDS engine (ASYNCcnt), then broadcast-read.
// ---------------------------------------------------------------------------
#define TJ 32
__global__ __launch_bounds__(128) void attn_core_kernel(
    const float* __restrict__ Q, const float* __restrict__ K,
    const float* __restrict__ V, const float* __restrict__ mask,
    const float* __restrict__ gamma, const float* __restrict__ alpha,
    _Float16* __restrict__ ctx_h) {
  __shared__ float qsh[128][DH + 1];
  __shared__ __align__(16) float ksh[TJ][DH];
  __shared__ __align__(16) float vsh[TJ][DH];
  __shared__ float msh[TJ];

  const int head = blockIdx.y;              // 0..7
  const int t    = threadIdx.x;             // 0..127
  const int i    = blockIdx.x * 128 + t;    // query row 0..511
  const size_t qoff = (size_t)i * DD + head * DH;

  const float scale = 1.0f / (gamma[0] * 8.0f);  // gamma * sqrt(64)
  const float a0    = alpha[0];

  // each thread loads its own q row (contiguous 64 floats)
#pragma unroll 4
  for (int d = 0; d < DH; ++d) qsh[t][d] = Q[qoff + d];

  float acc[DH];
#pragma unroll
  for (int d = 0; d < DH; ++d) acc[d] = 0.0f;

  for (int jc = 0; jc < S; jc += TJ) {
    __syncthreads();  // protect previous chunk's readers

    // Async global->LDS copy of the K/V chunk (TJ*DH floats = 8KB each).
    // 128 lanes x 4 rounds x 16B per matrix. LDS address = low 32 bits of
    // the generic pointer.
    {
      const float* Kbase = K + (size_t)jc * DD + head * DH;
      const float* Vbase = V + (size_t)jc * DD + head * DH;
#pragma unroll
      for (int u = 0; u < 4; ++u) {
        int f4 = t + u * 128;        // float4 index: 0..511
        int jj = f4 >> 4;            // row in chunk (DH/4 = 16 float4 per row)
        int d  = (f4 & 15) << 2;     // float offset within row (16B aligned)
        const float* gk = Kbase + (size_t)jj * DD + d;
        const float* gv = Vbase + (size_t)jj * DD + d;
        unsigned lk = (unsigned)(unsigned long long)(&ksh[jj][d]);
        unsigned lv = (unsigned)(unsigned long long)(&vsh[jj][d]);
        asm volatile("global_load_async_to_lds_b128 %0, %1, off"
                     :: "v"(lk), "v"(gk) : "memory");
        asm volatile("global_load_async_to_lds_b128 %0, %1, off"
                     :: "v"(lv), "v"(gv) : "memory");
      }
    }
    if (t < TJ) msh[t] = (1.0f - mask[jc + t]) * 1e10f;

    asm volatile("s_wait_asynccnt 0x0" ::: "memory");
    __syncthreads();

    for (int jj = 0; jj < TJ; ++jj) {
      float s = 0.0f;
#pragma unroll
      for (int d = 0; d < DH; ++d) s += fabsf(qsh[t][d] - ksh[jj][d]);
      float zp = fmaxf(s * scale - a0, 0.0f) + msh[jj];
#pragma unroll
      for (int d = 0; d < DH; ++d) acc[d] += fmaxf(vsh[jj][d] - zp, 0.0f);
    }
  }

#pragma unroll
  for (int d = 0; d < DH; ++d) ctx_h[qoff + d] = (_Float16)acc[d];
}

// ---------------------------------------------------------------------------
// Host side
// ---------------------------------------------------------------------------
extern "C" void kernel_launch(void* const* d_in, const int* in_sizes, int n_in,
                              void* d_out, int out_size, void* d_ws,
                              size_t ws_size, hipStream_t stream) {
  const float* X     = (const float*)d_in[0];
  const float* mask  = (const float*)d_in[1];
  const float* Wq    = (const float*)d_in[2];
  const float* bq    = (const float*)d_in[3];
  const float* Wk    = (const float*)d_in[4];
  const float* bk    = (const float*)d_in[5];
  const float* Wv    = (const float*)d_in[6];
  const float* bv    = (const float*)d_in[7];
  const float* Wo    = (const float*)d_in[8];
  const float* bo    = (const float*)d_in[9];
  const float* gamma = (const float*)d_in[10];
  const float* alpha = (const float*)d_in[11];
  float* out = (float*)d_out;

  // workspace layout (6 MB total)
  char* ws = (char*)d_ws;
  const size_t H16 = (size_t)S * DD * sizeof(_Float16);  // 512 KB
  const size_t F32 = (size_t)S * DD * sizeof(float);     // 1 MB
  _Float16* Xh  = (_Float16*)(ws + 0 * H16);
  _Float16* Wqt = (_Float16*)(ws + 1 * H16);
  _Float16* Wkt = (_Float16*)(ws + 2 * H16);
  _Float16* Wvt = (_Float16*)(ws + 3 * H16);
  _Float16* Wot = (_Float16*)(ws + 4 * H16);
  _Float16* Ch  = (_Float16*)(ws + 5 * H16);
  float* Qf = (float*)(ws + 6 * H16);
  float* Kf = (float*)(ws + 6 * H16 + F32);
  float* Vf = (float*)(ws + 6 * H16 + 2 * F32);

  const int n = S * DD;  // 262144
  cvt_f32_to_f16_kernel<<<n / 256, 256, 0, stream>>>(X, Xh, n);
  cvt_transpose_w_kernel<<<n / 256, 256, 0, stream>>>(Wq, Wqt);
  cvt_transpose_w_kernel<<<n / 256, 256, 0, stream>>>(Wk, Wkt);
  cvt_transpose_w_kernel<<<n / 256, 256, 0, stream>>>(Wv, Wvt);
  cvt_transpose_w_kernel<<<n / 256, 256, 0, stream>>>(Wo, Wot);

  gemm_qkv_kernel<<<dim3(128, 3), 256, 0, stream>>>(Xh, Wqt, Wkt, Wvt,
                                                    bq, bk, bv, Qf, Kf, Vf);

  attn_core_kernel<<<dim3(4, NH), 128, 0, stream>>>(Qf, Kf, Vf, mask,
                                                    gamma, alpha, Ch);

  gemm_out_kernel<<<128, 256, 0, stream>>>(Ch, Wot, bo, out);
}
